// TDNN_33191507264067
// MI455X (gfx1250) — matrix operand order repaired and checked by reading, not compiled, not run on hardware
//
#include <hip/hip_runtime.h>
#include <hip/hip_bf16.h>

typedef __attribute__((ext_vector_type(16))) __bf16 v16bf;
typedef __attribute__((ext_vector_type(8)))  __bf16 v8bf;
typedef __attribute__((ext_vector_type(2)))  __bf16 v2bf;
typedef __attribute__((ext_vector_type(8)))  float  v8f;
typedef __attribute__((ext_vector_type(4)))  float  v4f;
typedef __attribute__((ext_vector_type(4)))  int    v4i;
typedef __attribute__((ext_vector_type(4)))  unsigned int v4u;
typedef __attribute__((ext_vector_type(2)))  unsigned int v2u;

namespace {
constexpr int Bb = 32, Tt = 1024, Dd = 512, Oo = 512, Kk = 5;
constexpr int To = Tt - Kk + 1;          // 1020
constexpr int Mm = Bb * To;              // 32640 = 255 * 128
constexpr int BM = 128, BN = 128, BK = 32;
constexpr int NSTAGE = (Kk * Dd) / BK;   // 80 reduction steps of 32
constexpr int LDA = 40;                  // LDS row stride in ushorts (80B, padded)
}

#if __has_builtin(__builtin_amdgcn_global_load_async_to_lds_b128)
#define USE_ASYNC_LDS 1
typedef __attribute__((address_space(1))) v4i as1_v4i;   // global int4*
typedef __attribute__((address_space(3))) v4i as3_v4i;   // LDS int4*
__device__ __forceinline__ void async_copy_b128(const unsigned short* g,
                                                unsigned short* l) {
  __builtin_amdgcn_global_load_async_to_lds_b128((as1_v4i*)g, (as3_v4i*)l, 0, 0);
}
#define ASYNC_FENCE() asm volatile("s_wait_asynccnt 0x0" ::: "memory")
#else
#define USE_ASYNC_LDS 0
#define ASYNC_FENCE()
#endif

__device__ __forceinline__ unsigned short f2bf(float f) {
  unsigned u = __builtin_bit_cast(unsigned, f);
  u += 0x7FFFu + ((u >> 16) & 1u);
  return (unsigned short)(u >> 16);
}

__device__ __forceinline__ unsigned pack_bf16(float f0, float f1) {
#if __has_builtin(__builtin_amdgcn_cvt_pk_bf16_f32)
  v2bf p = __builtin_amdgcn_cvt_pk_bf16_f32(f0, f1);
  return __builtin_bit_cast(unsigned, p);
#else
  unsigned u0 = __builtin_bit_cast(unsigned, f0);
  unsigned u1 = __builtin_bit_cast(unsigned, f1);
  u0 += 0x7FFFu + ((u0 >> 16) & 1u);
  u1 += 0x7FFFu + ((u1 >> 16) & 1u);
  return (u0 >> 16) | (u1 & 0xFFFF0000u);
#endif
}

// Repack kernel[O][D][K] fp32 -> Wb[k][o][d] bf16 (one-time, 2.6 MB)
__global__ void prep_w_kernel(const float* __restrict__ w,
                              unsigned short* __restrict__ wb) {
  int i = blockIdx.x * blockDim.x + threadIdx.x;   // over Oo*Dd
  if (i >= Oo * Dd) return;
  int o = i / Dd, d = i - o * Dd;
  const float* src = w + ((size_t)o * Dd + d) * Kk;
#pragma unroll
  for (int k = 0; k < Kk; ++k)
    wb[((size_t)k * Oo + o) * Dd + d] = f2bf(src[k]);
}

// One-time x fp32 -> bf16 (33.5 MB); hoists ALL conversion out of the hot loop
__global__ void prep_x_kernel(const float* __restrict__ x,
                              unsigned short* __restrict__ xb) {
  int i = blockIdx.x * blockDim.x + threadIdx.x;   // over (Bb*Tt*Dd)/4
  v4f v = ((const v4f*)x)[i];
  v2u pk = {pack_bf16(v[0], v[1]), pack_bf16(v[2], v[3])};
  ((v2u*)xb)[i] = pk;
}

template <bool PRECONV>
__launch_bounds__(128)
__global__ void tdnn_wmma_kernel(const float* __restrict__ x,
                                 const unsigned short* __restrict__ xb,
                                 const unsigned short* __restrict__ wb,
                                 const float* __restrict__ bias,
                                 float* __restrict__ out) {
  __shared__ __attribute__((aligned(16))) unsigned short sA[2][BM * LDA];
  __shared__ __attribute__((aligned(16))) unsigned short sB[2][BN * LDA];
  __shared__ __attribute__((aligned(16))) float sC[4][16 * 20];

  const int tid     = threadIdx.x;       // 128 threads = 4 waves
  const int lane    = tid & 31;
  const int wid     = tid >> 5;
  const int wm      = wid & 1;           // wave m-offset * 64
  const int wn      = wid >> 1;          // wave n-offset * 64
  const int lane_lo = lane & 15;
  const int lane_hi = lane >> 4;

  const int tn = blockIdx.x;             // 0..3   (O tiles)
  const int tm = blockIdx.y;             // 0..254 (M tiles)

  // ---- B staging: 128 rows x 32 bf16; thread: 4 rows, one 16B chunk
  const int brow = tid >> 2;             // 0..31 (+32*j)
  const int bcol = (tid & 3) * 8;        // bf16 column 0/8/16/24
  const unsigned short* bbase[4];
#pragma unroll
  for (int j = 0; j < 4; ++j)
    bbase[j] = wb + ((size_t)(tn * BN + brow + 32 * j)) * Dd + bcol;

  // ---- A staging
  const float* abase[8];                 // fp32 path: 8 rows x one float4 chunk
  const unsigned short* axbase[4];       // bf16 path: 4 rows x one 16B chunk
  const int arow = tid >> 3;             // fp32 path row base
  const int acol = (tid & 7) * 4;
  if constexpr (PRECONV) {
#pragma unroll
    for (int j = 0; j < 4; ++j) {
      int r = tm * BM + brow + 32 * j;   // same row map as B
      int b = r / To;
      int t = r - b * To;
      axbase[j] = xb + ((size_t)(b * Tt + t)) * Dd + bcol;
    }
  } else {
#pragma unroll
    for (int j = 0; j < 8; ++j) {
      int r = tm * BM + arow + 16 * j;
      int b = r / To;
      int t = r - b * To;
      abase[j] = x + ((size_t)(b * Tt + t)) * Dd + acol;
    }
  }

  v4f areg[8];
#if !USE_ASYNC_LDS
  v4u breg[4];
  v4u axreg[4];
#endif

  // Issue stage-s staging for buffer `buf` (async path: pure DMA, no VGPRs)
  auto stage_issue = [&](int s, int buf) {
    const int off = (s >> 4) * Dd + (s & 15) * BK;       // k*D + d0 (elements)
#if USE_ASYNC_LDS
    if constexpr (PRECONV) {
#pragma unroll
      for (int j = 0; j < 4; ++j)
        async_copy_b128(axbase[j] + off,
                        &sA[buf][(brow + 32 * j) * LDA + bcol]);
    }
#pragma unroll
    for (int j = 0; j < 4; ++j)
      async_copy_b128(bbase[j] + off,
                      &sB[buf][(brow + 32 * j) * LDA + bcol]);
#else
    if constexpr (PRECONV) {
#pragma unroll
      for (int j = 0; j < 4; ++j)
        axreg[j] = *(const v4u*)(axbase[j] + off);
    }
#pragma unroll
    for (int j = 0; j < 4; ++j)
      breg[j] = *(const v4u*)(bbase[j] + off);
    (void)buf;
#endif
    if constexpr (!PRECONV) {
#pragma unroll
      for (int j = 0; j < 8; ++j)
        areg[j] = *(const v4f*)(abase[j] + off);
    }
  };

  // Register->LDS part of staging (fp32-A conversion and/or sync fallback)
  auto stage_commit = [&](int buf) {
    if constexpr (!PRECONV) {
#pragma unroll
      for (int j = 0; j < 8; ++j) {
        v2u pk = {pack_bf16(areg[j][0], areg[j][1]),
                  pack_bf16(areg[j][2], areg[j][3])};
        *(v2u*)&sA[buf][(arow + 16 * j) * LDA + acol] = pk;
      }
    }
#if !USE_ASYNC_LDS
    if constexpr (PRECONV) {
#pragma unroll
      for (int j = 0; j < 4; ++j)
        *(v4u*)&sA[buf][(brow + 32 * j) * LDA + bcol] = axreg[j];
    }
#pragma unroll
    for (int j = 0; j < 4; ++j)
      *(v4u*)&sB[buf][(brow + 32 * j) * LDA + bcol] = breg[j];
#else
    (void)buf;
#endif
  };

  v8f acc[4][4];
#pragma unroll
  for (int ms = 0; ms < 4; ++ms)
#pragma unroll
    for (int ns = 0; ns < 4; ++ns)
      acc[ms][ns] = (v8f){};

  auto compute = [&](int buf) {
    v16bf afrag[4], bfrag[4];
#pragma unroll
    for (int ms = 0; ms < 4; ++ms) {
      // A 16-bit 16x32 layout: lanes 0-15 row=lane K 0-7/16-23,
      //                        lanes 16-31 row=lane-16 K 8-15/24-31
      const unsigned short* p =
          &sA[buf][(wm * 64 + ms * 16 + lane_lo) * LDA + lane_hi * 8];
      ((v8bf*)&afrag[ms])[0] = *(const v8bf*)(p);
      ((v8bf*)&afrag[ms])[1] = *(const v8bf*)(p + 16);
    }
#pragma unroll
    for (int ns = 0; ns < 4; ++ns) {
      // B 32x16 layout: lanes 0-15 col=lane K 0-7/16-23; lanes 16-31 col K 8-15/24-31
      const unsigned short* p =
          &sB[buf][(wn * 64 + ns * 16 + lane_lo) * LDA + lane_hi * 8];
      ((v8bf*)&bfrag[ns])[0] = *(const v8bf*)(p);
      ((v8bf*)&bfrag[ns])[1] = *(const v8bf*)(p + 16);
    }
#pragma unroll
    for (int ms = 0; ms < 4; ++ms)
#pragma unroll
      for (int ns = 0; ns < 4; ++ns)
        acc[ms][ns] = __builtin_amdgcn_wmma_f32_16x16x32_bf16(
            false, afrag[ms], false, bfrag[ns], (short)0, acc[ms][ns],
            false, false);
  };

  // ---- main pipeline: double-buffered LDS, single barrier per step
  stage_issue(0, 0);
  stage_commit(0);
  ASYNC_FENCE();
  __syncthreads();

  for (int s = 0; s < NSTAGE; ++s) {
    const int pb = s & 1;
    if (s + 1 < NSTAGE) stage_issue(s + 1, 1 - pb);  // DMA overlaps compute
    if (s + 2 < NSTAGE) {
      const int off = ((s + 2) >> 4) * Dd + ((s + 2) & 15) * BK;
      if constexpr (PRECONV)
        __builtin_prefetch(axbase[0] + off, 0, 1);
      else
        __builtin_prefetch(abase[0] + off, 0, 1);
    }
    compute(pb);
    if (s + 1 < NSTAGE) stage_commit(1 - pb);
    ASYNC_FENCE();
    __syncthreads();
  }

  // ---- epilogue: bias + relu, per-wave LDS transpose, t-contiguous stores
  float* scr = sC[wid];
  const int obase_wave = tn * BN + wn * 64;
#pragma unroll
  for (int ms = 0; ms < 4; ++ms) {
#pragma unroll
    for (int ns = 0; ns < 4; ++ns) {
      const int og0 = obase_wave + ns * 16;
      const float bv = bias[og0 + lane_lo];
      // C layout: VGPR j -> m = j + 8*lane_hi, n = lane_lo
#pragma unroll
      for (int j = 0; j < 8; ++j) {
        float v = acc[ms][ns][j] + bv;
        v = v > 0.f ? v : 0.f;
        scr[lane_lo * 20 + (j + 8 * lane_hi)] = v;  // scr[o_row][t_col]
      }
      const int tbase = tm * BM + wm * 64 + ms * 16;
      // read back o-major, 4 consecutive t per lane (LDS in-order within wave)
#pragma unroll
      for (int p2 = 0; p2 < 2; ++p2) {
        const int orow = p2 * 8 + (lane >> 2);
        const int c    = (lane & 3) * 4;
        v4f v = *(const v4f*)&scr[orow * 20 + c];
        const int og = og0 + orow;
#pragma unroll
        for (int e = 0; e < 4; ++e) {
          const int r = tbase + c + e;
          const int b = r / To;
          const int t = r - b * To;
          out[((size_t)b * Oo + og) * To + t] = v[e];
        }
      }
    }
  }
}

extern "C" void kernel_launch(void* const* d_in, const int* in_sizes, int n_in,
                              void* d_out, int out_size, void* d_ws, size_t ws_size,
                              hipStream_t stream) {
  const float* x    = (const float*)d_in[0];   // [32,1024,512] f32
  const float* w    = (const float*)d_in[1];   // [512,512,5]   f32
  const float* bias = (const float*)d_in[2];   // [512]         f32
  float* out        = (float*)d_out;           // [32,512,1020] f32

  const size_t welems = (size_t)Kk * Oo * Dd;        // 1,310,720
  const size_t xelems = (size_t)Bb * Tt * Dd;        // 16,777,216
  unsigned short* wb = (unsigned short*)d_ws;        // bf16 weights

  prep_w_kernel<<<(Oo * Dd + 255) / 256, 256, 0, stream>>>(w, wb);

  dim3 grid(Oo / BN, Mm / BM);   // (4, 255)
  if (ws_size >= (welems + xelems) * sizeof(unsigned short)) {
    // Full path: bf16 x in workspace, main loop stages A and B via async DMA
    unsigned short* xxb = wb + welems;               // 16B-aligned offset
    prep_x_kernel<<<(int)(xelems / 4 / 256), 256, 0, stream>>>(x, xxb);
    tdnn_wmma_kernel<true><<<grid, 128, 0, stream>>>(x, xxb, wb, bias, out);
  } else {
    // Fallback: convert x on the fly inside the GEMM kernel
    tdnn_wmma_kernel<false><<<grid, 128, 0, stream>>>(x, nullptr, wb, bias, out);
  }
}
